// PAGATv2_60335700574564
// MI455X (gfx1250) — matrix operand order repaired and checked by reading, not compiled, not run on hardware
//
#include <hip/hip_runtime.h>
#include <hip/hip_bf16.h>

typedef __attribute__((ext_vector_type(16))) __bf16 v16bf;
typedef __attribute__((ext_vector_type(8)))  __bf16 v8bf;
typedef __attribute__((ext_vector_type(8)))  float  v8f;

#define GAT_H   4
#define GAT_C   32
#define GAT_HC  128
#define NEG_SLOPE 0.2f

// ---------- float <-> order-preserving uint (for atomicMax on f32 scores) ----
__device__ __forceinline__ unsigned mapf(float f) {
    unsigned u = __float_as_uint(f);
    return (u & 0x80000000u) ? ~u : (u | 0x80000000u);
}
__device__ __forceinline__ float unmapf(unsigned u) {
    return (u & 0x80000000u) ? __uint_as_float(u & 0x7FFFFFFFu)
                             : __uint_as_float(~u);
}

// ---------- prep: transpose + convert a 128x128 f32 weight to bf16 ----------
__global__ __launch_bounds__(256) void prep_w(const float* __restrict__ W,
                                              __bf16* __restrict__ Wt) {
    int i = blockIdx.x * 256 + threadIdx.x;      // 16384 threads
    int k = i >> 7, n = i & 127;
    Wt[(size_t)n * GAT_HC + k] = (__bf16)W[(size_t)k * GAT_HC + n];
}

// ---------- convert f32 vector -> bf16 --------------------------------------
__global__ __launch_bounds__(256) void conv_bf(const float* __restrict__ x,
                                               __bf16* __restrict__ xb, int n) {
    int i = blockIdx.x * 256 + threadIdx.x;
    if (i < n) xb[i] = (__bf16)x[i];
}

// ---------- dual WMMA GEMM: XL = Xbf @ Wl, XR = Xbf @ Wr --------------------
// X: [M,128] bf16 row-major. WlT/WrT: [128,128] bf16, WT[n][k] = W[k][n].
// One workgroup = 8 waves = one 16-row block; wave w owns output cols 16w..16w+15
// and computes BOTH Wl and Wr tiles, reusing its A fragments.
__global__ __launch_bounds__(256) void gemm_dual(const __bf16* __restrict__ X,
                                                 const __bf16* __restrict__ WlT,
                                                 const __bf16* __restrict__ WrT,
                                                 float* __restrict__ XL,
                                                 float* __restrict__ XR, int M) {
    const int wave = threadIdx.x >> 5;
    const int lane = threadIdx.x & 31;
    const int r    = lane & 15;
    const int hi   = lane >> 4;
    const int row0 = blockIdx.x * 16;
    const int n0   = wave * 16;

    int arow = row0 + r; if (arow >= M) arow = M - 1;      // clamp (M mult of 16 anyway)
    const __bf16* A  = X   + (size_t)arow * GAT_HC;
    const __bf16* Bl = WlT + (size_t)(n0 + r) * GAT_HC;    // row n of W^T = col n of W
    const __bf16* Br = WrT + (size_t)(n0 + r) * GAT_HC;

    v8f cl = {}; v8f cr = {};
#pragma unroll
    for (int k = 0; k < GAT_HC; k += 32) {
        // A fragment 16x32 bf16: elems 0..7 -> K = k + hi*16 + 0..7
        //                        elems 8..15 -> K = k + hi*16 + 8 + 0..7
        v8bf a0 = *(const v8bf*)(A + k + hi * 16);
        v8bf a1 = *(const v8bf*)(A + k + hi * 16 + 8);
        v16bf av;
#pragma unroll
        for (int i = 0; i < 8; ++i) { av[i] = a0[i]; av[i + 8] = a1[i]; }

        // B fragment 32x16 bf16: lane col = r, K = k + hi*16 + 0..15 (contiguous in W^T row)
        v8bf bl0 = *(const v8bf*)(Bl + k + hi * 16);
        v8bf bl1 = *(const v8bf*)(Bl + k + hi * 16 + 8);
        v16bf blv;
#pragma unroll
        for (int i = 0; i < 8; ++i) { blv[i] = bl0[i]; blv[i + 8] = bl1[i]; }

        v8bf br0 = *(const v8bf*)(Br + k + hi * 16);
        v8bf br1 = *(const v8bf*)(Br + k + hi * 16 + 8);
        v16bf brv;
#pragma unroll
        for (int i = 0; i < 8; ++i) { brv[i] = br0[i]; brv[i + 8] = br1[i]; }

        cl = __builtin_amdgcn_wmma_f32_16x16x32_bf16(false, av, false, blv,
                                                     (short)0, cl, false, false);
        cr = __builtin_amdgcn_wmma_f32_16x16x32_bf16(false, av, false, brv,
                                                     (short)0, cr, false, false);
    }
    // C/D layout: VGPR i -> row m = i + 8*hi, col = n0 + r
#pragma unroll
    for (int i = 0; i < 8; ++i) {
        int m = row0 + i + 8 * hi;
        if (m < M) {
            XL[(size_t)m * GAT_HC + n0 + r] = cl[i];
            XR[(size_t)m * GAT_HC + n0 + r] = cr[i];
        }
    }
}

// ---------- pass 1: per-edge scores + segment max ---------------------------
// one wave per edge; lane = channel-in-head; heads unrolled
__global__ __launch_bounds__(256) void edge_score(const long long* __restrict__ src,
                                                  const long long* __restrict__ dst,
                                                  const float* __restrict__ ea,
                                                  const float* __restrict__ xl,
                                                  const float* __restrict__ xr,
                                                  const float* __restrict__ We,
                                                  const float* __restrict__ att,
                                                  float* __restrict__ score,
                                                  unsigned* __restrict__ smax, int E) {
    int e = blockIdx.x * 8 + (threadIdx.x >> 5);
    if (e >= E) return;
    int lane = threadIdx.x & 31;
    long long s = src[e], d = dst[e];
    float eav = ea[e];
    const float* xls = xl + (size_t)s * GAT_HC;
    const float* xrd = xr + (size_t)d * GAT_HC;

    float sc[GAT_H];
#pragma unroll
    for (int h = 0; h < GAT_H; ++h) {
        int c = h * GAT_C + lane;
        float m = xls[c] + xrd[c] + eav * We[c];
        m = (m > 0.f) ? m : NEG_SLOPE * m;     // leaky_relu
        sc[h] = m * att[c];
    }
#pragma unroll
    for (int off = 16; off > 0; off >>= 1) {
#pragma unroll
        for (int h = 0; h < GAT_H; ++h) sc[h] += __shfl_xor(sc[h], off, 32);
    }
    if (lane < GAT_H) {
        float v = sc[lane];
        score[(size_t)e * GAT_H + lane] = v;
        atomicMax(&smax[(size_t)d * GAT_H + lane], mapf(v));
    }
}

// ---------- pass 2: ex = exp(score - smax[dst]); denom += ex ----------------
__global__ __launch_bounds__(256) void edge_exp(const long long* __restrict__ dst,
                                                float* __restrict__ score,
                                                const unsigned* __restrict__ smax,
                                                float* __restrict__ denom, int EH) {
    int i = blockIdx.x * 256 + threadIdx.x;    // over E*H
    if (i >= EH) return;
    int e = i >> 2, h = i & 3;
    long long d = dst[e];
    float mx = unmapf(smax[(size_t)d * GAT_H + h]);
    float ex = __expf(score[i] - mx);
    score[i] = ex;
    unsafeAtomicAdd(&denom[(size_t)d * GAT_H + h], ex);
}

// ---------- pass 3: agg[dst] += (ex/denom[dst]) * xl[src] -------------------
__global__ __launch_bounds__(256) void edge_agg(const long long* __restrict__ src,
                                                const long long* __restrict__ dst,
                                                const float* __restrict__ ex,
                                                const float* __restrict__ denom,
                                                const float* __restrict__ xl,
                                                float* __restrict__ agg, int E) {
    int e = blockIdx.x * 8 + (threadIdx.x >> 5);
    if (e >= E) return;
    int lane = threadIdx.x & 31;
    long long s = src[e], d = dst[e];
    const float* xls = xl + (size_t)s * GAT_HC;
    float* ag = agg + (size_t)d * GAT_HC;
#pragma unroll
    for (int h = 0; h < GAT_H; ++h) {
        float alpha = ex[(size_t)e * GAT_H + h] / denom[(size_t)d * GAT_H + h];
        int c = h * GAT_C + lane;
        unsafeAtomicAdd(&ag[c], alpha * xls[c]);
    }
}

// ---------- layer-1 epilogue: h = elu(agg + b1) -> bf16 ---------------------
__global__ __launch_bounds__(256) void node_post1(const float* __restrict__ agg,
                                                  const float* __restrict__ b1,
                                                  __bf16* __restrict__ hb, int n) {
    int i = blockIdx.x * 256 + threadIdx.x;
    if (i >= n) return;
    float v = agg[i] + b1[i & (GAT_HC - 1)];
    v = (v > 0.f) ? v : (__expf(v) - 1.0f);
    hb[i] = (__bf16)v;
}

// ---------- final: head-mean + b2, elu, @Wo + bo, sigmoid -------------------
__global__ __launch_bounds__(256) void finalize(const float* __restrict__ agg,
                                                const float* __restrict__ b2,
                                                const float* __restrict__ Wo,
                                                const float* __restrict__ bo,
                                                float* __restrict__ out, int Nn) {
    int n = blockIdx.x * 8 + (threadIdx.x >> 5);
    if (n >= Nn) return;
    int lane = threadIdx.x & 31;
    const float* a = agg + (size_t)n * GAT_HC;
    float mean = 0.25f * (a[lane] + a[lane + 32] + a[lane + 64] + a[lane + 96]);
    float v = mean + b2[lane];
    v = (v > 0.f) ? v : (__expf(v) - 1.0f);
    float p = v * Wo[lane];
#pragma unroll
    for (int off = 16; off > 0; off >>= 1) p += __shfl_xor(p, off, 32);
    if (lane == 0) out[n] = 1.0f / (1.0f + __expf(-(p + bo[0])));
}

extern "C" void kernel_launch(void* const* d_in, const int* in_sizes, int n_in,
                              void* d_out, int out_size, void* d_ws, size_t ws_size,
                              hipStream_t stream) {
    const float*     x   = (const float*)d_in[0];
    const long long* ei  = (const long long*)d_in[1];   // int64 [2,E]
    const float*     ea  = (const float*)d_in[2];
    const float* Wl1 = (const float*)d_in[3];
    const float* Wr1 = (const float*)d_in[4];
    const float* We1 = (const float*)d_in[5];
    const float* a1  = (const float*)d_in[6];
    const float* b1  = (const float*)d_in[7];
    const float* Wl2 = (const float*)d_in[8];
    const float* Wr2 = (const float*)d_in[9];
    const float* We2 = (const float*)d_in[10];
    const float* a2  = (const float*)d_in[11];
    const float* b2  = (const float*)d_in[12];
    const float* Wo  = (const float*)d_in[13];
    const float* bo  = (const float*)d_in[14];

    const int Nn = in_sizes[0] / GAT_HC;   // 50000
    const int E  = in_sizes[2];            // 800000
    const long long* src = ei;
    const long long* dst = ei + E;

    char* p = (char*)d_ws;
    auto take = [&](size_t bytes) -> char* {
        char* r = p; p += (bytes + 255) & ~(size_t)255; return r;
    };
    __bf16*   xbf   = (__bf16*)take((size_t)Nn * GAT_HC * sizeof(__bf16));
    __bf16*   wbf   = (__bf16*)take((size_t)4 * GAT_HC * GAT_HC * sizeof(__bf16));
    float*    xl    = (float*)take((size_t)Nn * GAT_HC * sizeof(float));
    float*    xr    = (float*)take((size_t)Nn * GAT_HC * sizeof(float));
    float*    sc    = (float*)take((size_t)E * GAT_H * sizeof(float));
    unsigned* smax  = (unsigned*)take((size_t)Nn * GAT_H * sizeof(unsigned));
    float*    denom = (float*)take((size_t)Nn * GAT_H * sizeof(float));
    float*    agg   = (float*)take((size_t)Nn * GAT_HC * sizeof(float));

    __bf16* Wl1t = wbf;
    __bf16* Wr1t = wbf + 1 * GAT_HC * GAT_HC;
    __bf16* Wl2t = wbf + 2 * GAT_HC * GAT_HC;
    __bf16* Wr2t = wbf + 3 * GAT_HC * GAT_HC;

    // ---- prep: weights -> bf16 transposed; x -> bf16 ----
    prep_w<<<64, 256, 0, stream>>>(Wl1, Wl1t);
    prep_w<<<64, 256, 0, stream>>>(Wr1, Wr1t);
    prep_w<<<64, 256, 0, stream>>>(Wl2, Wl2t);
    prep_w<<<64, 256, 0, stream>>>(Wr2, Wr2t);
    conv_bf<<<(Nn * GAT_HC + 255) / 256, 256, 0, stream>>>(x, xbf, Nn * GAT_HC);

    const int gemmGrid = (Nn + 15) / 16;
    const int edgeGrid = (E + 7) / 8;
    const int nodeGrid = (Nn + 7) / 8;

    // ================= layer 1 =================
    hipMemsetAsync(smax, 0, (size_t)Nn * GAT_H * sizeof(unsigned), stream);
    hipMemsetAsync(denom, 0, (size_t)Nn * GAT_H * sizeof(float), stream);
    hipMemsetAsync(agg, 0, (size_t)Nn * GAT_HC * sizeof(float), stream);

    gemm_dual<<<gemmGrid, 256, 0, stream>>>(xbf, Wl1t, Wr1t, xl, xr, Nn);
    edge_score<<<edgeGrid, 256, 0, stream>>>(src, dst, ea, xl, xr, We1, a1, sc, smax, E);
    edge_exp<<<(E * GAT_H + 255) / 256, 256, 0, stream>>>(dst, sc, smax, denom, E * GAT_H);
    edge_agg<<<edgeGrid, 256, 0, stream>>>(src, dst, sc, denom, xl, agg, E);
    node_post1<<<(Nn * GAT_HC + 255) / 256, 256, 0, stream>>>(agg, b1, xbf, Nn * GAT_HC);

    // ================= layer 2 =================
    hipMemsetAsync(smax, 0, (size_t)Nn * GAT_H * sizeof(unsigned), stream);
    hipMemsetAsync(denom, 0, (size_t)Nn * GAT_H * sizeof(float), stream);
    hipMemsetAsync(agg, 0, (size_t)Nn * GAT_HC * sizeof(float), stream);

    gemm_dual<<<gemmGrid, 256, 0, stream>>>(xbf, Wl2t, Wr2t, xl, xr, Nn);
    edge_score<<<edgeGrid, 256, 0, stream>>>(src, dst, ea, xl, xr, We2, a2, sc, smax, E);
    edge_exp<<<(E * GAT_H + 255) / 256, 256, 0, stream>>>(dst, sc, smax, denom, E * GAT_H);
    edge_agg<<<edgeGrid, 256, 0, stream>>>(src, dst, sc, denom, xl, agg, E);

    finalize<<<nodeGrid, 256, 0, stream>>>(agg, b2, Wo, bo, (float*)d_out, Nn);
}
